// Network_1400159338850
// MI455X (gfx1250) — matrix operand order repaired and checked by reading, not compiled, not run on hardware
//
#include <hip/hip_runtime.h>
#include <hip/hip_bf16.h>

// ---------------------------------------------------------------------------
// Types for CDNA5 WMMA (gfx1250, wave32)
// ---------------------------------------------------------------------------
typedef __attribute__((ext_vector_type(16))) __bf16 v16bf;
typedef __attribute__((ext_vector_type(8)))  __bf16 v8bf;
typedef __attribute__((ext_vector_type(8)))  float  v8f;

union BFrag { v16bf f; v8bf h[2]; };

// float -> bf16 (round to nearest even), stored as raw u16
static __device__ __forceinline__ unsigned short f2bf(float f) {
    unsigned int u = __float_as_uint(f);
    u = (u + 0x7FFFu + ((u >> 16) & 1u)) >> 16;
    return (unsigned short)u;
}

// Loihi CUBA constants
#define AI_DECAY 0.75f       /* (4096-1024)/4096 */
#define AV_DECAY 0.96875f    /* (4096-128)/4096  */
#define THETA    5120.0f     /* 80 * 64          */
#define WSCALE   64.0f

// ---------------------------------------------------------------------------
// Elementwise f32 -> bf16 conversion (weights)
// ---------------------------------------------------------------------------
__global__ __launch_bounds__(256) void convert_f32_bf16(
    const float* __restrict__ in, unsigned short* __restrict__ out, int n) {
    for (int i = blockIdx.x * 256 + threadIdx.x; i < n; i += gridDim.x * 256)
        out[i] = f2bf(in[i]);
}

// ---------------------------------------------------------------------------
// x[b][i][t] (f32) -> xb[b][t][i] (bf16) tiled transpose (32x32 via LDS)
// ---------------------------------------------------------------------------
__global__ __launch_bounds__(256) void transpose_x_bf16(
    const float* __restrict__ x, unsigned short* __restrict__ xb,
    int Nin, int T) {
    __shared__ float tile[32][33];
    const int b  = blockIdx.z;
    const int t0 = blockIdx.x * 32;
    const int i0 = blockIdx.y * 32;
    const int tx = threadIdx.x;       // 0..31
    const int ty = threadIdx.y;       // 0..7
    #pragma unroll
    for (int j = 0; j < 32; j += 8) {
        int i = i0 + ty + j;
        tile[ty + j][tx] = x[((size_t)b * Nin + i) * T + t0 + tx]; // coalesced in t
    }
    __syncthreads();
    #pragma unroll
    for (int j = 0; j < 32; j += 8) {
        int t = t0 + ty + j;
        xb[((size_t)b * T + t) * Nin + i0 + tx] = f2bf(tile[tx][ty + j]); // coalesced in i
    }
}

// ---------------------------------------------------------------------------
// Fused WMMA-GEMM + Loihi CUBA neuron scan (+1-step axonal delay).
//
// One workgroup owns (batch b = blockIdx.y, 128-channel block n0 = 128*blockIdx.x)
// and walks the full T sequentially in chunks of 128 timesteps:
//   1) WMMA:  tile[128 t][128 ch] = A[b, t0..t0+127, :K] . W[n0..n0+127, :K]^T
//      (8 waves, each a 32t x 64ch patch = 2x4 chain of v_wmma_f32_16x16x32_bf16)
//   2) stage tile in LDS (stride 129 -> bank-conflict-free in every phase)
//   3) threads 0..127 advance the sequential neuron recurrence 128 steps,
//      carrying cur/vol/sprev in registers across chunks.
//
// MODE 0 (layer 1): emit bf16 spikes to s1[b][t][ch]  (Nstride = Nhid),
//                   which is exactly layer-2's A-operand layout.
// MODE 1 (layer 2): emit f32 spikes to out[b][ch][t]  (Nstride = Nout),
//                   via an in-LDS transpose + cooperative coalesced store.
// ---------------------------------------------------------------------------
template <int MODE>
__global__ __launch_bounds__(256) void fused_gemm_neuron(
    const __bf16* __restrict__ A,   // [b][t][k], row stride K (bf16)
    const __bf16* __restrict__ W,   // [n][k] (bf16)
    void* __restrict__ outp,
    int T, int K, int Nstride) {
    __shared__ float tile[128][129];

    const int tid  = threadIdx.x;
    const int lane = tid & 31;
    const int wid  = tid >> 5;
    const int wm   = wid & 3;                 // 4 waves along t
    const int wn   = wid >> 2;                // 2 waves along ch
    const int lr   = lane & 15;
    const int half = lane >> 4;
    const int b    = blockIdx.y;
    const int n0   = blockIdx.x * 128;

    const __bf16* aPanel = A + ((size_t)b * T) * K;
    const __bf16* wR[4];
    #pragma unroll
    for (int nt = 0; nt < 4; ++nt)
        wR[nt] = W + (size_t)(n0 + wn * 64 + nt * 16 + lr) * K;

    // per-channel neuron state (threads 0..127 own channel n0+tid)
    float cur = 0.f, vol = 0.f, sprev = 0.f;

    for (int t0 = 0; t0 < T; t0 += 128) {
        // Prefetch next chunk's A panel (contiguous 128*K bf16) into caches.
        if (t0 + 128 < T) {
            const char* nxt = (const char*)(aPanel + (size_t)(t0 + 128) * K);
            const size_t bytes = (size_t)128 * K * sizeof(__bf16);
            for (size_t off = (size_t)tid * 128; off < bytes; off += 256u * 128u)
                __builtin_prefetch(nxt + off, 0, 0);   // global_prefetch_b8
        }

        // ---- WMMA phase: 128t x 128ch x K ----
        v8f acc[2][4] = {};
        const __bf16* aR[2];
        aR[0] = aPanel + (size_t)(t0 + wm * 32 + lr) * K;
        aR[1] = aR[0] + (size_t)16 * K;

        for (int k0 = 0; k0 < K; k0 += 32) {
            BFrag aF[2], bF[4];
            // A 16x32 frag: lanes 0-15 K={0..7,16..23}, lanes 16-31 K={8..15,24..31}
            #pragma unroll
            for (int mt = 0; mt < 2; ++mt) {
                aF[mt].h[0] = *(const v8bf*)(aR[mt] + k0 + half * 8);
                aF[mt].h[1] = *(const v8bf*)(aR[mt] + k0 + 16 + half * 8);
            }
            // B 32x16 frag: lanes 0-15 K={0..7,16..23} of column n, lanes 16-31 K={8..15,24..31}
            #pragma unroll
            for (int nt = 0; nt < 4; ++nt) {
                bF[nt].h[0] = *(const v8bf*)(wR[nt] + k0 + half * 8);
                bF[nt].h[1] = *(const v8bf*)(wR[nt] + k0 + 16 + half * 8);
            }
            #pragma unroll
            for (int mt = 0; mt < 2; ++mt)
                #pragma unroll
                for (int nt = 0; nt < 4; ++nt)
                    acc[mt][nt] = __builtin_amdgcn_wmma_f32_16x16x32_bf16(
                        false, aF[mt].f, false, bF[nt].f,
                        (short)0, acc[mt][nt], false, false);
        }

        // ---- Stage C tile to LDS: tile[t_local][ch_local] ----
        #pragma unroll
        for (int mt = 0; mt < 2; ++mt)
            #pragma unroll
            for (int nt = 0; nt < 4; ++nt)
                #pragma unroll
                for (int j = 0; j < 8; ++j)
                    tile[wm * 32 + mt * 16 + half * 8 + j]
                        [wn * 64 + nt * 16 + lr] = acc[mt][nt][j];
        __syncthreads();

        // ---- Neuron scan phase (sequential in t; threads 0..127) ----
        if (MODE == 0) {
            if (tid < 128) {
                unsigned short* s1 = (unsigned short*)outp;
                const size_t cidx = (size_t)n0 + tid;
                for (int t = 0; t < 128; ++t) {
                    float xv = tile[t][tid];
                    cur = AI_DECAY * cur + WSCALE * xv;
                    vol = AV_DECAY * vol + cur;
                    s1[((size_t)b * T + t0 + t) * Nstride + cidx] =
                        (sprev != 0.f) ? (unsigned short)0x3F80u : (unsigned short)0u;
                    float sp = (vol >= THETA) ? 1.f : 0.f;
                    vol *= (1.f - sp);
                    sprev = sp;
                }
            }
            __syncthreads();   // protect tile before next chunk overwrites it
        } else {
            if (tid < 128) {
                for (int t = 0; t < 128; ++t) {
                    float xv = tile[t][tid];
                    cur = AI_DECAY * cur + WSCALE * xv;
                    vol = AV_DECAY * vol + cur;
                    tile[t][tid] = sprev;            // delayed spike at t0+t
                    float sp = (vol >= THETA) ? 1.f : 0.f;
                    vol *= (1.f - sp);
                    sprev = sp;
                }
            }
            __syncthreads();
            // Cooperative transpose store: out[b][ch][t], coalesced over t.
            float* outF = (float*)outp;
            for (int e = tid; e < 128 * 128; e += 256) {
                int ch = e >> 7, t = e & 127;
                outF[((size_t)(b * Nstride + n0 + ch)) * T + t0 + t] = tile[t][ch];
            }
            __syncthreads();
        }
    }
}

// ---------------------------------------------------------------------------
// Launch: convert weights, transpose x, fused layer 1, fused layer 2.
// ---------------------------------------------------------------------------
extern "C" void kernel_launch(void* const* d_in, const int* in_sizes, int n_in,
                              void* d_out, int out_size, void* d_ws, size_t ws_size,
                              hipStream_t stream) {
    (void)in_sizes; (void)n_in; (void)out_size; (void)ws_size;
    const int B = 64, Nin = 512, Nhid = 1024, Nout = 512, T = 1024;

    const float* x  = (const float*)d_in[0];     // (B, Nin, T)
    const float* w1 = (const float*)d_in[1];     // (Nhid, Nin)
    const float* w2 = (const float*)d_in[2];     // (Nout, Nhid)
    float* out = (float*)d_out;                  // (B, Nout, T)

    // Workspace layout (all segment sizes are multiples of 256B)
    unsigned short* xb  = (unsigned short*)d_ws;                 // B*T*Nin   bf16
    unsigned short* w1b = xb  + (size_t)B * T * Nin;             // Nhid*Nin  bf16
    unsigned short* w2b = w1b + (size_t)Nhid * Nin;              // Nout*Nhid bf16
    unsigned short* s1  = w2b + (size_t)Nout * Nhid;             // B*T*Nhid  bf16

    // Weight conversion
    convert_f32_bf16<<<512, 256, 0, stream>>>(w1, w1b, Nhid * Nin);
    convert_f32_bf16<<<512, 256, 0, stream>>>(w2, w2b, Nout * Nhid);

    // x -> [b][t][i] bf16
    {
        dim3 tb(32, 8, 1);
        dim3 tg(T / 32, Nin / 32, B);
        transpose_x_bf16<<<tg, tb, 0, stream>>>(x, xb, Nin, T);
    }

    // Layer 1 fused: s1[b][t][hid] (bf16 spikes, delayed 1)
    {
        dim3 g(Nhid / 128, B, 1);
        fused_gemm_neuron<0><<<g, 256, 0, stream>>>(
            (const __bf16*)xb, (const __bf16*)w1b, (void*)s1, T, Nin, Nhid);
    }

    // Layer 2 fused: out[b][n][t] (f32 spikes, delayed 1)
    {
        dim3 g(Nout / 128, B, 1);
        fused_gemm_neuron<1><<<g, 256, 0, stream>>>(
            (const __bf16*)s1, (const __bf16*)w2b, (void*)out, T, Nhid, Nout);
    }
}
